// Attention_1657857376451
// MI455X (gfx1250) — matrix operand order repaired and checked by reading, not compiled, not run on hardware
//
#include <hip/hip_runtime.h>

// Fused attention for MI455X (gfx1250, wave32, WMMA).
// d_out = [out (B*H*S*D f32)] ++ [p_attn (B*H*S*S f32)].
// d_ws: needs 3 * B*H*S*D * 2 bytes = 24 MiB for bf16 copies of Q(pre-scaled),K,V.

typedef unsigned int u32;
typedef unsigned short u16;

typedef __attribute__((ext_vector_type(16))) __bf16 v16bf;
typedef __attribute__((ext_vector_type(8)))  float  v8f;
typedef __attribute__((ext_vector_type(8)))  u32    v8u;
typedef __attribute__((ext_vector_type(4)))  float  f4v;

#define NB 2
#define NH 16
#define SS 2048
#define DD 64

constexpr float kNegInf = -1.0e9f;
constexpr float kScale  = 0.125f;  // 1/sqrt(64)

// ---- LDS layout (bytes), all 16B aligned ----
constexpr int SC_STRIDE  = 2052;     // f32 per score row (2048 + 4 pad -> conflict-free lane->bank)
constexpr int OFF_SC     = 0;        // 16*2052*4           = 131328
constexpr int OFF_QS     = 131328;   // 16 rows * 36 u32    =   2304 -> 133632 (Q tile bf16, padded)
constexpr int OFF_VST    = 133632;   // 64 cols * 136 u16   =  17408 -> 151040 (V 128-row chunk, transposed)
constexpr int OFF_RED    = 151040;   // 16*16 f32           =   1024 -> 152064
constexpr int OFF_STAT   = 152064;   // 32 f32 (max, 1/sum) =    128 -> 152192
constexpr int OFF_PACC   = 152192;   // 2*16*64 f32         =   8192 -> 160384
constexpr int SMEM_BYTES = 160384;   // 2 WGs/WGP: 320768 < 327680

__device__ __forceinline__ u32 pack2bf(float a, float b) {
  // round-to-nearest-even f32 -> bf16, packed pair
  u32 x = __float_as_uint(a);
  u32 y = __float_as_uint(b);
  x = x + 0x7FFFu + ((x >> 16) & 1u);
  y = y + 0x7FFFu + ((y >> 16) & 1u);
  return (x >> 16) | (y & 0xFFFF0000u);
}

// ---- pre-pass: f32 -> bf16 (Q pre-scaled by 1/sqrt(D)) ----
__global__ void __launch_bounds__(256) prep_bf16(
    const float* __restrict__ Q, const float* __restrict__ K,
    const float* __restrict__ V, u32* __restrict__ qb,
    u32* __restrict__ kb, u32* __restrict__ vb, int n2) {
  int i = blockIdx.x * 256 + threadIdx.x;
  if (i >= n2) return;
  const float2* Q2 = (const float2*)Q;
  const float2* K2 = (const float2*)K;
  const float2* V2 = (const float2*)V;
  float2 q = Q2[i];
  qb[i] = pack2bf(q.x * kScale, q.y * kScale);
  float2 k = K2[i];
  kb[i] = pack2bf(k.x, k.y);
  float2 v = V2[i];
  vb[i] = pack2bf(v.x, v.y);
}

// ---- fused attention: one workgroup per (b*h, 16-row q tile) ----
__global__ void __launch_bounds__(256) attn_fused(
    const u32* __restrict__ qb, const u32* __restrict__ kb,
    const u32* __restrict__ vb, const int* __restrict__ mask,
    float* __restrict__ out_o, float* __restrict__ out_p) {
  extern __shared__ char smem[];
  float* sc   = (float*)(smem + OFF_SC);    // [16][2052] masked/scaled scores -> probs
  u32*   QsU  = (u32*)(smem + OFF_QS);      // [16][36] u32 view of bf16 Q tile
  u32*   VstU = (u32*)(smem + OFF_VST);     // [64 col][68 u32] transposed V chunk (bf16 row pairs)
  float* red  = (float*)(smem + OFF_RED);   // [16][16] reduction scratch
  float* stat = (float*)(smem + OFF_STAT);  // [0:16) rowmax, [16:32) 1/rowsum
  float* pacc = (float*)(smem + OFF_PACC);  // [2][16][64] PV partials

  const int t    = threadIdx.x;
  const int lane = t & 31;
  const int wv   = t >> 5;     // wave 0..7
  const int n    = lane & 15;  // N-col of C / row-in-tile for A
  const int hi   = lane >> 4;  // lane half
  const int qt   = blockIdx.x; // 0..127
  const int bh   = blockIdx.y; // 0..31
  const int bsel = bh >> 4;    // batch (H=16)
  const int q0   = qt * 16;

  const u32* qbh = qb + (size_t)bh * (SS * DD / 2);
  const u32* kbh = kb + (size_t)bh * (SS * DD / 2);
  const u32* vbh = vb + (size_t)bh * (SS * DD / 2);

  // ---- stage Q tile (16 x 64 bf16 = 512 u32) into padded LDS ----
  for (int it = 0; it < 2; ++it) {
    int lin = it * 256 + t;            // 0..511
    int m = lin >> 5, c = lin & 31;
    QsU[m * 36 + c] = qbh[(size_t)(q0 + m) * 32 + c];
  }
  __syncthreads();

  // ---- A fragments (Q), kept in VGPRs for the whole QK^T phase ----
  // A 16x32 bf16 layout: lanes0-15 K={0..7,16..23}, lanes16-31 K={8..15,24..31}
  const uint4* Q4 = (const uint4*)QsU;
  uint4 A0a = Q4[n * 9 + hi];
  uint4 A0b = Q4[n * 9 + 2 + hi];
  uint4 A1a = Q4[n * 9 + 4 + hi];
  uint4 A1b = Q4[n * 9 + 6 + hi];
  v8u au0 = {A0a.x, A0a.y, A0a.z, A0a.w, A0b.x, A0b.y, A0b.z, A0b.w};
  v8u au1 = {A1a.x, A1a.y, A1a.z, A1a.w, A1b.x, A1b.y, A1b.z, A1b.w};
  v16bf a0 = __builtin_bit_cast(v16bf, au0);
  v16bf a1 = __builtin_bit_cast(v16bf, au1);

  // ---- QK^T (D=64 = 2 WMMA steps), mask, write scores to LDS ----
  for (int kt = wv; kt < SS / 16; kt += 8) {
    int col0 = kt * 16;
    int krow = col0 + n;
    // B[d][n] = K[krow=col0+n][d]: per step K={8h..8h+7, 16+8h..16+8h+7} (matches A)
    const uint4* kp4 = (const uint4*)(kbh + (size_t)krow * 32 + hi * 4);
    uint4 B0a = kp4[0], B0b = kp4[2], B1a = kp4[4], B1b = kp4[6];
    v8u bu0 = {B0a.x, B0a.y, B0a.z, B0a.w, B0b.x, B0b.y, B0b.z, B0b.w};
    v8u bu1 = {B1a.x, B1a.y, B1a.z, B1a.w, B1b.x, B1b.y, B1b.z, B1b.w};
    v16bf b0 = __builtin_bit_cast(v16bf, bu0);
    v16bf b1 = __builtin_bit_cast(v16bf, bu1);
    v8f c = {};
    c = __builtin_amdgcn_wmma_f32_16x16x32_bf16(false, a0, false, b0, (short)0, c, false, false);
    c = __builtin_amdgcn_wmma_f32_16x16x32_bf16(false, a1, false, b1, (short)0, c, false, false);
    const int* mrow = mask + (size_t)bsel * SS * SS + (size_t)q0 * SS + col0 + n;
#pragma unroll
    for (int r = 0; r < 8; ++r) {            // C: VGPR r -> M = r + 8*hi
      int m = r + hi * 8;
      int mv = mrow[(size_t)m * SS];
      sc[m * SC_STRIDE + col0 + n] = mv ? c[r] : kNegInf;
    }
  }
  __syncthreads();

  // ---- softmax over 2048 cols per row (16 threads per row) ----
  const int sr = t >> 4, sj = t & 15;
  {
    float pm = -3.0e38f;
#pragma unroll 8
    for (int i = 0; i < 128; ++i)
      pm = fmaxf(pm, sc[sr * SC_STRIDE + sj + 16 * i]);
    red[sr * 16 + sj] = pm;
  }
  __syncthreads();
  if (t < 16) {
    float m = red[t * 16];
#pragma unroll
    for (int i = 1; i < 16; ++i) m = fmaxf(m, red[t * 16 + i]);
    stat[t] = m;
  }
  __syncthreads();
  {
    float rm = stat[sr];
    float s = 0.f;
#pragma unroll 4
    for (int i = 0; i < 128; ++i) {
      int idx = sr * SC_STRIDE + sj + 16 * i;
      float e = __expf(sc[idx] - rm);
      sc[idx] = e;
      s += e;
    }
    red[sr * 16 + sj] = s;
  }
  __syncthreads();
  if (t < 16) {
    float s = red[t * 16];
#pragma unroll
    for (int i = 1; i < 16; ++i) s += red[t * 16 + i];
    stat[16 + t] = 1.0f / s;
  }
  __syncthreads();

  // ---- normalize in LDS + stream p_attn (write-only -> non-temporal) ----
  float* pdst = out_p + ((size_t)bh * SS + q0) * SS;  // contiguous 16*2048 block
  for (int it = 0; it < 32; ++it) {
    int f4 = it * 256 + t;   // 8192 float4s
    int rr = f4 >> 9;
    int c4 = f4 & 511;
    f4v v = *(f4v*)&sc[rr * SC_STRIDE + c4 * 4];
    v *= stat[16 + rr];
    *(f4v*)&sc[rr * SC_STRIDE + c4 * 4] = v;
    __builtin_nontemporal_store(v, (f4v*)pdst + (size_t)rr * 512 + c4);
  }

  // ---- PV: out[16x64] = P(16x2048) @ V(2048x64) ----
  // wave -> (vt = 16-col slab of D, ks = 64-row half of each 128-row V chunk)
  const int vt = wv & 3, ks = wv >> 2;
  const int cn = vt * 16 + n;
  const uint4* V4 = (const uint4*)VstU;
  v8f acc = {};
  for (int ch = 0; ch < 16; ++ch) {
    __syncthreads();
    // stage V chunk (128 rows x 64 cols) transposed: VstU[col][rowpair]
    for (int it = 0; it < 8; ++it) {
      int lin = it * 256 + t;            // 2048 row-pair x col-pair units
      int rr2 = lin >> 5, c2 = lin & 31; // row pair (2rr2,2rr2+1), col pair (2c2,2c2+1)
      u32 w0 = vbh[(size_t)(ch * 128 + 2 * rr2) * 32 + c2];
      u32 w1 = vbh[(size_t)(ch * 128 + 2 * rr2 + 1) * 32 + c2];
      VstU[(2 * c2) * 68 + rr2]     = (w0 & 0xFFFFu) | (w1 << 16);
      VstU[(2 * c2 + 1) * 68 + rr2] = (w0 >> 16) | (w1 & 0xFFFF0000u);
    }
    __syncthreads();
#pragma unroll
    for (int s = 0; s < 2; ++s) {        // two K=32 steps over this wave's 64-row half
      // A fragment: P rows from LDS (f32 -> bf16)
      int k0 = ch * 128 + ks * 64 + s * 32 + hi * 8;
      const float* arow = &sc[n * SC_STRIDE + k0];
      v8u au;
#pragma unroll
      for (int v2 = 0; v2 < 4; ++v2) {
        au[v2]     = pack2bf(arow[2 * v2],      arow[2 * v2 + 1]);
        au[4 + v2] = pack2bf(arow[16 + 2 * v2], arow[16 + 2 * v2 + 1]);
      }
      v16bf av = __builtin_bit_cast(v16bf, au);
      // B fragment: contiguous, aligned b128 reads from transposed V
      uint4 Bva = V4[cn * 17 + ks * 8 + s * 4 + hi];
      uint4 Bvb = V4[cn * 17 + ks * 8 + s * 4 + hi + 2];
      v8u bu = {Bva.x, Bva.y, Bva.z, Bva.w, Bvb.x, Bvb.y, Bvb.z, Bvb.w};
      v16bf bv = __builtin_bit_cast(v16bf, bu);
      acc = __builtin_amdgcn_wmma_f32_16x16x32_bf16(false, av, false, bv, (short)0, acc, false, false);
    }
  }

  // ---- deterministic cross-wave reduction (2 k-halves per vt) ----
#pragma unroll
  for (int r = 0; r < 8; ++r)
    pacc[ks * 1024 + (r + hi * 8) * 64 + cn] = acc[r];
  __syncthreads();
  f4v o0 = *(f4v*)&pacc[t * 4];
  f4v o1 = *(f4v*)&pacc[1024 + t * 4];
  f4v osum = o0 + o1;
  float* odst = out_o + ((size_t)bh * SS + q0) * DD;  // contiguous 16*64 block
  *((f4v*)odst + t) = osum;
}

extern "C" void kernel_launch(void* const* d_in, const int* in_sizes, int n_in,
                              void* d_out, int out_size, void* d_ws, size_t ws_size,
                              hipStream_t stream) {
  (void)in_sizes; (void)n_in; (void)out_size; (void)ws_size;
  const float* Q   = (const float*)d_in[0];
  const float* K   = (const float*)d_in[1];
  const float* V   = (const float*)d_in[2];
  const int* mask  = (const int*)d_in[3];
  float* out_o = (float*)d_out;
  float* out_p = out_o + (size_t)NB * NH * SS * DD;

  const size_t n_el = (size_t)NB * NH * SS * DD;  // 4,194,304 per tensor
  u32* qb = (u32*)d_ws;                           // ws: 3 * 8 MiB bf16 copies
  u32* kb = qb + n_el / 2;
  u32* vb = kb + n_el / 2;
  int n2 = (int)(n_el / 2);

  prep_bf16<<<(n2 + 255) / 256, 256, 0, stream>>>(Q, K, V, qb, kb, vb, n2);

  hipFuncSetAttribute(reinterpret_cast<const void*>(attn_fused),
                      hipFuncAttributeMaxDynamicSharedMemorySize, SMEM_BYTES);
  dim3 grid(SS / 16, NB * NH);
  attn_fused<<<grid, 256, SMEM_BYTES, stream>>>(qb, kb, vb, mask, out_o, out_p);
}
